// GPT2_14113262534965
// MI455X (gfx1250) — matrix operand order repaired and checked by reading, not compiled, not run on hardware
//
#include <hip/hip_runtime.h>
#include <hip/hip_bf16.h>

// ---------------- types ----------------
typedef __attribute__((ext_vector_type(16))) __bf16 v16bf;
typedef __attribute__((ext_vector_type(8)))  __bf16 v8bf;
typedef __attribute__((ext_vector_type(2)))  __bf16 v2bf;
typedef __attribute__((ext_vector_type(8)))  float  v8f;
typedef __attribute__((ext_vector_type(2)))  float  v2f;

#define L_   12
#define B_   4
#define S_   1024
#define D_   1024
#define H_   16
#define HD_  64
#define DF_  4096
#define TOK  (B_ * S_)      // 4096 rows
#define NEG_BIG (-3.0e38f)

// fp32 -> bf16 (RNE) via native conversion; backend may use v_cvt_pk_bf16_f32
__device__ __forceinline__ __bf16 to_bf16(float f) { return (__bf16)f; }

// pack two floats -> u32 with lo in [15:0], hi in [31:16] (both RNE bf16)
__device__ __forceinline__ unsigned pack_bf16(float lo, float hi) {
    v2f x; x[0] = lo; x[1] = hi;
    v2bf r = __builtin_convertvector(x, v2bf);
    return __builtin_bit_cast(unsigned, r);
}

__device__ __forceinline__ v16bf cat8(v8bf lo, v8bf hi) {
    v16bf r;
#pragma unroll
    for (int i = 0; i < 8; ++i) { r[i] = lo[i]; r[8 + i] = hi[i]; }
    return r;
}

// branch-free gelu_new: 0.5*x*(1+tanh(t)) with tanh(t) = 1 - 2/(1+exp(2t))
__device__ __forceinline__ float gelu_new(float x) {
    const float c = 0.7978845608028654f; // sqrt(2/pi)
    float t = c * (x + 0.044715f * x * x * x);
    float e = __expf(2.0f * t);          // inf/0 saturate tanh to +/-1 cleanly
    return 0.5f * x * (2.0f - 2.0f / (1.0f + e));
}

// ---------------- embedding: h = wte[ids] + wpe[s] (fp32 residual) ----------------
__global__ __launch_bounds__(256)
void embed_kernel(const int* __restrict__ ids, const float* __restrict__ wte,
                  const float* __restrict__ wpe, float* __restrict__ h) {
    int t = blockIdx.x;
    int s = t & (S_ - 1);
    int id = ids[t];
#pragma unroll
    for (int i = 0; i < D_ / 256; ++i) {
        int c = threadIdx.x + i * 256;
        h[(size_t)t * D_ + c] = wte[(size_t)id * D_ + c] + wpe[(size_t)s * D_ + c];
    }
}

// ---------------- layernorm: fp32 in -> bf16 (or fp32) out ----------------
__global__ __launch_bounds__(256)
void ln_kernel(const float* __restrict__ x, const float* __restrict__ g,
               const float* __restrict__ b, __bf16* __restrict__ yb,
               float* __restrict__ yf) {
    __shared__ float s1[256], s2[256];
    int t = blockIdx.x;
    int c = threadIdx.x * 4;               // 4 consecutive elements per thread
    float4 v = *(const float4*)(x + (size_t)t * D_ + c);
    float sum = v.x + v.y + v.z + v.w;
    float sq  = v.x * v.x + v.y * v.y + v.z * v.z + v.w * v.w;
    s1[threadIdx.x] = sum; s2[threadIdx.x] = sq;
    __syncthreads();
    for (int off = 128; off > 0; off >>= 1) {
        if (threadIdx.x < off) {
            s1[threadIdx.x] += s1[threadIdx.x + off];
            s2[threadIdx.x] += s2[threadIdx.x + off];
        }
        __syncthreads();
    }
    float mean = s1[0] * (1.0f / D_);
    float var  = s2[0] * (1.0f / D_) - mean * mean;
    float rstd = rsqrtf(var + 1e-5f);
    float4 gg = *(const float4*)(g + c);
    float4 bb = *(const float4*)(b + c);
    float y0 = (v.x - mean) * rstd * gg.x + bb.x;
    float y1 = (v.y - mean) * rstd * gg.y + bb.y;
    float y2 = (v.z - mean) * rstd * gg.z + bb.z;
    float y3 = (v.w - mean) * rstd * gg.w + bb.w;
    if (yf) {
        float4 o; o.x = y0; o.y = y1; o.z = y2; o.w = y3;
        *(float4*)(yf + (size_t)t * D_ + c) = o;
    } else {
        uint2 u; u.x = pack_bf16(y0, y1); u.y = pack_bf16(y2, y3);
        *(uint2*)(yb + (size_t)t * D_ + c) = u;
    }
}

// ---------------- WMMA bf16 GEMM ----------------
// A: [M,K] bf16 row-major (activations). Bw: [K,N] f32 row-major (weights,
// converted to bf16 in-flight). out: fp32 (+residual) or bf16 (+gelu).
// 256 threads, tile 128x128, K-step 32, 8 waves in 2(M)x4(N),
// double-buffered LDS + register prefetch, one barrier per k-step.
__global__ __launch_bounds__(256)
void gemm_wmma(const __bf16* __restrict__ A, const float* __restrict__ Bw,
               const float* __restrict__ bias, const float* __restrict__ res,
               float* __restrict__ outf, __bf16* __restrict__ outb,
               int M, int N, int K, int do_gelu) {
    __shared__ __align__(32) __bf16 As[2][128][32];   // [m][k]
    __shared__ __align__(32) __bf16 Bs[2][128][32];   // transposed: [n][k]

    const int tid  = threadIdx.x;
    const int wave = tid >> 5;
    const int lane = tid & 31;
    const int hf   = lane >> 4;
    const int l16  = lane & 15;
    const int wm   = wave >> 2;
    const int wn   = wave & 3;
    const int mblk = blockIdx.y * 128;
    const int nblk = blockIdx.x * 128;

    // A slots: pure bf16 copy. rows a_row, a_row+64; 8 elems at a_c8.
    const int a_row = tid >> 2, a_c8 = (tid & 3) * 8;
    // B slots: k-row pairs (2*kp, 2*kp+1) at 4 cols b_n4; kp and kp+8.
    const int b_kp = tid >> 5, b_n4 = (tid & 31) * 4;

    v8bf  aR[2];
    float4 bR0[2], bR1[2];

    auto load_tile = [&](int k0) {
#pragma unroll
        for (int i = 0; i < 2; ++i) {
            aR[i] = *(const v8bf*)(A + (size_t)(mblk + a_row + i * 64) * K + k0 + a_c8);
            int kk = k0 + 2 * (b_kp + i * 8);
            bR0[i] = *(const float4*)(Bw + (size_t)kk * N + nblk + b_n4);
            bR1[i] = *(const float4*)(Bw + (size_t)(kk + 1) * N + nblk + b_n4);
        }
    };
    auto store_tile = [&](int buf) {
#pragma unroll
        for (int i = 0; i < 2; ++i) {
            *(v8bf*)&As[buf][a_row + i * 64][a_c8] = aR[i];
            int k2 = 2 * (b_kp + i * 8);
            *(unsigned*)&Bs[buf][b_n4 + 0][k2] = pack_bf16(bR0[i].x, bR1[i].x);
            *(unsigned*)&Bs[buf][b_n4 + 1][k2] = pack_bf16(bR0[i].y, bR1[i].y);
            *(unsigned*)&Bs[buf][b_n4 + 2][k2] = pack_bf16(bR0[i].z, bR1[i].z);
            *(unsigned*)&Bs[buf][b_n4 + 3][k2] = pack_bf16(bR0[i].w, bR1[i].w);
        }
    };

    v8f acc[4][2];
#pragma unroll
    for (int r = 0; r < 4; ++r)
#pragma unroll
        for (int f = 0; f < 2; ++f)
#pragma unroll
            for (int j = 0; j < 8; ++j) acc[r][f][j] = 0.0f;

    const int nk = K >> 5;
    load_tile(0);
    for (int kt = 0; kt < nk; ++kt) {
        const int buf = kt & 1;
        store_tile(buf);
        __syncthreads();
        if (kt + 1 < nk) load_tile((kt + 1) << 5);   // hidden behind WMMAs
        if (kt + 2 < nk) {                           // global_prefetch_b8 of weights
            int k2 = ((kt + 2) << 5) + 2 * b_kp;
            __builtin_prefetch(Bw + (size_t)k2 * N + nblk + b_n4, 0, 1);
        }

        v16bf afr[4], bfr[2];
#pragma unroll
        for (int r = 0; r < 4; ++r) {
            int row = wm * 64 + r * 16 + l16;
            v8bf lo = *(const v8bf*)&As[buf][row][hf * 16];
            v8bf hi = *(const v8bf*)&As[buf][row][hf * 16 + 8];
            afr[r] = cat8(lo, hi);
        }
#pragma unroll
        for (int f = 0; f < 2; ++f) {
            int n = wn * 32 + f * 16 + l16;
            bfr[f] = *(const v16bf*)&Bs[buf][n][hf * 16];
        }
#pragma unroll
        for (int r = 0; r < 4; ++r)
#pragma unroll
            for (int f = 0; f < 2; ++f)
                acc[r][f] = __builtin_amdgcn_wmma_f32_16x16x32_bf16(
                    false, afr[r], false, bfr[f], (short)0, acc[r][f], false, false);
        // next iteration's barrier separates buffer reuse
    }

    // ---- epilogue ----
#pragma unroll
    for (int r = 0; r < 4; ++r)
#pragma unroll
        for (int f = 0; f < 2; ++f) {
            int n = nblk + wn * 32 + f * 16 + l16;
            float bn = bias[n];
#pragma unroll
            for (int j = 0; j < 8; ++j) {
                int m = mblk + wm * 64 + r * 16 + hf * 8 + j;
                float v = acc[r][f][j] + bn;
                if (do_gelu) v = gelu_new(v);
                if (outf) {
                    if (res) v += res[(size_t)m * N + n];
                    outf[(size_t)m * N + n] = v;
                } else {
                    outb[(size_t)m * N + n] = to_bf16(v);
                }
            }
        }
}

// ---------------- flash attention (causal), WMMA bf16, bf16 qkv ----------------
// grid: (S/64, H, B), block 128 (4 waves x 16 q-rows). qkv: [tok][3*D] bf16.
__global__ __launch_bounds__(128)
void attn_kernel(const __bf16* __restrict__ qkv, __bf16* __restrict__ out) {
    __shared__ __align__(32) __bf16 Qs[64][64];
    __shared__ __align__(32) __bf16 Ks[2][32][64];      // [key][hd]
    __shared__ __align__(32) __bf16 Vt[2][64][32];      // [hd][key]
    __shared__ __align__(32) __bf16 Ps[4][16][32];      // per-wave P tile

    const int tid  = threadIdx.x;
    const int wave = tid >> 5;
    const int lane = tid & 31;
    const int hf   = lane >> 4;
    const int l16  = lane & 15;
    const int qt   = blockIdx.x;
    const int head = blockIdx.y;
    const int b    = blockIdx.z;
    const int qbase = qt * 64;
    const size_t tokbase = (size_t)b * S_;
    const int stride3 = 3 * D_;
    const float scale = 0.125f;  // 1/sqrt(64), applied to scores post-WMMA

    // stage Q: pure bf16 copy, 512 v8bf over 128 threads (4 each)
#pragma unroll
    for (int i = 0; i < 4; ++i) {
        int idx = i * 128 + tid;
        int row = idx >> 3, c8 = (idx & 7) * 8;
        *(v8bf*)&Qs[row][c8] =
            *(const v8bf*)(qkv + (tokbase + qbase + row) * stride3 + head * HD_ + c8);
    }

    // K/V slots: keys skey, skey+16; 8 elems at sc8
    const int skey = tid >> 3, sc8 = (tid & 7) * 8;
    v8bf kR[2], vR[2];
    auto load_kv = [&](int kb) {
#pragma unroll
        for (int i = 0; i < 2; ++i) {
            const size_t base = (tokbase + kb * 32 + skey + i * 16) * (size_t)stride3
                                + head * HD_ + sc8;
            kR[i] = *(const v8bf*)(qkv + base + D_);
            vR[i] = *(const v8bf*)(qkv + base + 2 * D_);
        }
    };
    auto store_kv = [&](int buf) {
#pragma unroll
        for (int i = 0; i < 2; ++i) {
            int key = skey + i * 16;
            *(v8bf*)&Ks[buf][key][sc8] = kR[i];
#pragma unroll
            for (int e = 0; e < 8; ++e) Vt[buf][sc8 + e][key] = vR[i][e];
        }
    };

    __syncthreads();   // Qs visible

    v16bf qa[2];
#pragma unroll
    for (int c = 0; c < 2; ++c) {
        int row = wave * 16 + l16;
        v8bf lo = *(const v8bf*)&Qs[row][c * 32 + hf * 16];
        v8bf hi = *(const v8bf*)&Qs[row][c * 32 + hf * 16 + 8];
        qa[c] = cat8(lo, hi);
    }

    float m_i[8], l_i[8];
    v8f O[4];
#pragma unroll
    for (int j = 0; j < 8; ++j) { m_i[j] = NEG_BIG; l_i[j] = 0.0f; }
#pragma unroll
    for (int n = 0; n < 4; ++n)
#pragma unroll
        for (int j = 0; j < 8; ++j) O[n][j] = 0.0f;

    const int nkb = 2 * qt + 2;
    load_kv(0);
    for (int kb = 0; kb < nkb; ++kb) {
        const int buf = kb & 1;
        store_kv(buf);
        __syncthreads();
        if (kb + 1 < nkb) load_kv(kb + 1);

        // scores: 16(q) x 32(key) in two 16x16 C-fragments
        v8f sc[2];
#pragma unroll
        for (int f = 0; f < 2; ++f) {
#pragma unroll
            for (int j = 0; j < 8; ++j) sc[f][j] = 0.0f;
#pragma unroll
            for (int c = 0; c < 2; ++c) {
                int key = f * 16 + l16;
                v16bf kb16 = *(const v16bf*)&Ks[buf][key][c * 32 + hf * 16];
                sc[f] = __builtin_amdgcn_wmma_f32_16x16x32_bf16(
                    false, qa[c], false, kb16, (short)0, sc[f], false, false);
            }
        }

        // scale + causal mask (C layout: col = l16, row = hf*8 + j)
        int qg0 = qbase + wave * 16 + hf * 8;
#pragma unroll
        for (int f = 0; f < 2; ++f) {
            int kg = kb * 32 + f * 16 + l16;
#pragma unroll
            for (int j = 0; j < 8; ++j) {
                float s = sc[f][j] * scale;
                sc[f][j] = (kg > qg0 + j) ? NEG_BIG : s;
            }
        }

        // online softmax per row (rows live across 16-lane halves)
        float p0[8], p1[8];
#pragma unroll
        for (int j = 0; j < 8; ++j) {
            float rmax = fmaxf(sc[0][j], sc[1][j]);
#pragma unroll
            for (int msk = 1; msk < 16; msk <<= 1)
                rmax = fmaxf(rmax, __shfl_xor(rmax, msk, 16));
            float newm = fmaxf(m_i[j], rmax);
            float fac = __expf(m_i[j] - newm);
            p0[j] = __expf(sc[0][j] - newm);
            p1[j] = __expf(sc[1][j] - newm);
            float rsum = p0[j] + p1[j];
#pragma unroll
            for (int msk = 1; msk < 16; msk <<= 1)
                rsum += __shfl_xor(rsum, msk, 16);
            l_i[j] = l_i[j] * fac + rsum;
            m_i[j] = newm;
#pragma unroll
            for (int n = 0; n < 4; ++n) O[n][j] *= fac;
        }

        // P: C-layout -> LDS -> A-layout (per-wave region, in-wave DS order)
#pragma unroll
        for (int j = 0; j < 8; ++j) {
            Ps[wave][hf * 8 + j][l16]      = to_bf16(p0[j]);
            Ps[wave][hf * 8 + j][16 + l16] = to_bf16(p1[j]);
        }
        {
            v8bf lo = *(const v8bf*)&Ps[wave][l16][hf * 16];
            v8bf hi = *(const v8bf*)&Ps[wave][l16][hf * 16 + 8];
            v16bf pa = cat8(lo, hi);
#pragma unroll
            for (int n = 0; n < 4; ++n) {
                v16bf vb = *(const v16bf*)&Vt[buf][n * 16 + l16][hf * 16];
                O[n] = __builtin_amdgcn_wmma_f32_16x16x32_bf16(
                    false, pa, false, vb, (short)0, O[n], false, false);
            }
        }
    }

    // normalize and write merged heads (bf16): out[tok][head*64 + hd]
    float invl[8];
#pragma unroll
    for (int j = 0; j < 8; ++j) invl[j] = 1.0f / l_i[j];
#pragma unroll
    for (int n = 0; n < 4; ++n)
#pragma unroll
        for (int j = 0; j < 8; ++j) {
            int q = qbase + wave * 16 + hf * 8 + j;
            int hd = n * 16 + l16;
            out[(tokbase + q) * D_ + head * HD_ + hd] = to_bf16(O[n][j] * invl[j]);
        }
}

// ---------------- host orchestration ----------------
extern "C" void kernel_launch(void* const* d_in, const int* in_sizes, int n_in,
                              void* d_out, int out_size, void* d_ws, size_t ws_size,
                              hipStream_t stream) {
    (void)in_sizes; (void)n_in; (void)out_size; (void)ws_size;
    const int*   ids   = (const int*)  d_in[0];
    const float* wte   = (const float*)d_in[1];
    const float* wpe   = (const float*)d_in[2];
    const float* ln1_g = (const float*)d_in[3];
    const float* ln1_b = (const float*)d_in[4];
    const float* wqkv  = (const float*)d_in[5];
    const float* bqkv  = (const float*)d_in[6];
    const float* wproj = (const float*)d_in[7];
    const float* bproj = (const float*)d_in[8];
    const float* ln2_g = (const float*)d_in[9];
    const float* ln2_b = (const float*)d_in[10];
    const float* wfc   = (const float*)d_in[11];
    const float* bfc   = (const float*)d_in[12];
    const float* wfc2  = (const float*)d_in[13];
    const float* bfc2  = (const float*)d_in[14];
    const float* lnf_g = (const float*)d_in[15];
    const float* lnf_b = (const float*)d_in[16];

    float*  h    = (float*)d_ws;                         // TOK*D fp32 residual
    __bf16* xln  = (__bf16*)(h + (size_t)TOK * D_);      // TOK*D bf16
    __bf16* qkvb = xln  + (size_t)TOK * D_;              // TOK*3D bf16
    __bf16* attb = qkvb + (size_t)TOK * 3 * D_;          // TOK*D bf16
    __bf16* afc  = attb + (size_t)TOK * D_;              // TOK*DF bf16

    embed_kernel<<<TOK, 256, 0, stream>>>(ids, wte, wpe, h);

    for (int l = 0; l < L_; ++l) {
        ln_kernel<<<TOK, 256, 0, stream>>>(h, ln1_g + l * D_, ln1_b + l * D_,
                                           xln, nullptr);

        gemm_wmma<<<dim3(3 * D_ / 128, TOK / 128), 256, 0, stream>>>(
            xln, wqkv + (size_t)l * D_ * 3 * D_, bqkv + (size_t)l * 3 * D_,
            nullptr, nullptr, qkvb, TOK, 3 * D_, D_, 0);

        attn_kernel<<<dim3(S_ / 64, H_, B_), 128, 0, stream>>>(qkvb, attb);

        gemm_wmma<<<dim3(D_ / 128, TOK / 128), 256, 0, stream>>>(
            attb, wproj + (size_t)l * D_ * D_, bproj + (size_t)l * D_,
            h, h, nullptr, TOK, D_, D_, 0);

        ln_kernel<<<TOK, 256, 0, stream>>>(h, ln2_g + l * D_, ln2_b + l * D_,
                                           xln, nullptr);

        gemm_wmma<<<dim3(DF_ / 128, TOK / 128), 256, 0, stream>>>(
            xln, wfc + (size_t)l * D_ * DF_, bfc + (size_t)l * DF_,
            nullptr, nullptr, afc, TOK, DF_, D_, 1);

        gemm_wmma<<<dim3(D_ / 128, TOK / 128), 256, 0, stream>>>(
            afc, wfc2 + (size_t)l * DF_ * D_, bfc2 + (size_t)l * D_,
            h, h, nullptr, TOK, D_, DF_, 0);
    }

    ln_kernel<<<TOK, 256, 0, stream>>>(h, lnf_g, lnf_b, nullptr, (float*)d_out);
}
